// GraphSage_5274219840014
// MI455X (gfx1250) — compile-verified, matrix operand order checked by reading, not run on hardware
//
#include <hip/hip_runtime.h>

typedef unsigned short u16;
typedef unsigned int   u32;
typedef __bf16 bf16_t;
typedef __attribute__((ext_vector_type(16))) bf16_t v16bf;
typedef __attribute__((ext_vector_type(8)))  float  v8f;

#define FEAT  1433
#define HID   128
#define BATCH 2048
#define SAMP  11
#define NL1   (BATCH * SAMP)   /* 22528 */
#define K1    (2 * FEAT)       /* 2866  */
#define K1PAD 2880
#define K2    (2 * HID)        /* 256   */

// ---------- bf16 helpers (RNE) ----------
__device__ __forceinline__ u16 f2bf(float f) {
  union { float f; u32 u; } v; v.f = f;
  u32 u = v.u;
  u32 r = u + 0x7FFFu + ((u >> 16) & 1u);
  return (u16)(r >> 16);
}
__device__ __forceinline__ float bf2f(u16 h) {
  union { u32 u; float f; } v; v.u = ((u32)h) << 16;
  return v.f;
}

union FragBF { u32 u[8]; v16bf v; };

__device__ __forceinline__ v8f wmma_bf16(const FragBF& a, const FragBF& b, v8f c) {
  return __builtin_amdgcn_wmma_f32_16x16x32_bf16(false, a.v, false, b.v,
                                                 (short)0, c, false, false);
}

// 16-bit A/B fragment layout: lane (m, m+16) split; VGPR j holds K pair at
// kj = (j<4 ? 2j : 2j+8) + (lane>=16 ? 8 : 0)   (dword-aligned pairs).
__device__ __forceinline__ int frag_k(int j, int hi) {
  return ((j < 4) ? 2 * j : 2 * j + 8) + hi * 8;
}

// Repack W1 [128, 2866] f32 -> bf16, layout [K/32 chunk][o 0..127][k%32], K padded to 2880
__global__ void prep_w1(const float* __restrict__ w, u16* __restrict__ wt) {
  int idx = blockIdx.x * 256 + threadIdx.x;
  if (idx >= 128 * K1PAD) return;
  int o = idx / K1PAD, k = idx - o * K1PAD;
  float v = (k < K1) ? w[o * K1 + k] : 0.0f;
  wt[(k >> 5) * (128 * 32) + o * 32 + (k & 31)] = f2bf(v);
}

// Repack W2 [128, 256] f32 -> bf16, same chunked layout
__global__ void prep_w2(const float* __restrict__ w, u16* __restrict__ wt) {
  int idx = blockIdx.x * 256 + threadIdx.x;
  if (idx >= 128 * K2) return;
  int o = idx / K2, k = idx - o * K2;
  wt[(k >> 5) * (128 * 32) + o * 32 + (k & 31)] = f2bf(w[o * K2 + k]);
}

// ---------------- Layer 1: fused gather/mean + bf16 WMMA GEMM + ReLU ----------------
// grid = NL1/32 blocks, 128 threads (4 waves). Block: 32-row m-tile x 128 outputs.
// Double-buffered A tile in LDS (one barrier per K chunk); B fragments straight
// from L2-resident repacked weights, reused across both A sub-tiles.
__global__ __launch_bounds__(128) void sage_layer1(
    const float* __restrict__ raw, const int* __restrict__ nb1,
    const int* __restrict__ neigh1, const u16* __restrict__ w1t,
    u16* __restrict__ h1bf) {
  __shared__ u16 lds_a[2][32 * 32];
  __shared__ int lds_nb[32];
  __shared__ int lds_ne[32 * SAMP];

  const int tid   = threadIdx.x;
  const int wave  = tid >> 5;
  const int lane  = tid & 31;
  const int mbase = blockIdx.x * 32;

  if (tid < 32) lds_nb[tid] = nb1[mbase + tid];
  for (int i = tid; i < 32 * SAMP; i += 128)
    lds_ne[i] = neigh1[(size_t)(mbase + i / SAMP) * SAMP + (i % SAMP)];

  const int hi   = (lane >= 16) ? 1 : 0;
  const int mrow = lane & 15;
  const int col0 = wave * 32 + (lane & 15);
  const int col1 = col0 + 16;

  auto stage = [&](int buf, int kb) {
    u16* dst = lds_a[buf];
    for (int e = tid; e < 1024; e += 128) {
      int row = e >> 5;
      int kk  = e & 31;
      int k   = kb + kk;
      float v = 0.0f;
      if (k < FEAT) {
        v = raw[(size_t)lds_nb[row] * FEAT + k];
      } else if (k < K1) {
        int ka = k - FEAT;
        const int* ne = &lds_ne[row * SAMP];
        float s = 0.0f;
#pragma unroll
        for (int j = 0; j < SAMP; ++j)
          s += raw[(size_t)ne[j] * FEAT + ka];
        v = s * (1.0f / SAMP);
      }
      dst[row * 32 + kk] = f2bf(v);
    }
  };

  v8f acc00 = {}, acc01 = {}, acc10 = {}, acc11 = {};

  __syncthreads();        // indices visible
  stage(0, 0);
  __syncthreads();        // buf0 ready

  constexpr int NC = K1PAD / 32;  // 90
  for (int c = 0; c < NC; ++c) {
    const int kb  = c * 32;
    const int cur = c & 1;
    const u16* abuf = lds_a[cur];
    const u16* wchunk = w1t + (size_t)kb * 128;   // 128*32 bf16 chunk block

    FragBF a0, a1, b0, b1;
#pragma unroll
    for (int j = 0; j < 8; ++j) {
      int kj = frag_k(j, hi);
      a0.u[j] = *(const u32*)&abuf[mrow * 32 + kj];
      a1.u[j] = *(const u32*)&abuf[(16 + mrow) * 32 + kj];
      b0.u[j] = *(const u32*)&wchunk[col0 * 32 + kj];
      b1.u[j] = *(const u32*)&wchunk[col1 * 32 + kj];
    }
    acc00 = wmma_bf16(a0, b0, acc00);
    acc01 = wmma_bf16(a0, b1, acc01);
    acc10 = wmma_bf16(a1, b0, acc10);
    acc11 = wmma_bf16(a1, b1, acc11);

    if (c + 1 < NC) stage(cur ^ 1, kb + 32);   // overlap next gather with WMMA
    __syncthreads();
  }

  // D layout: VGPR r -> M = r + hi*8, N = lane&15. ReLU then bf16 store.
#pragma unroll
  for (int r = 0; r < 8; ++r) {
    int m0 = mbase + r + hi * 8;
    int m1 = m0 + 16;
    float v00 = acc00[r]; v00 = v00 > 0.0f ? v00 : 0.0f;
    float v01 = acc01[r]; v01 = v01 > 0.0f ? v01 : 0.0f;
    float v10 = acc10[r]; v10 = v10 > 0.0f ? v10 : 0.0f;
    float v11 = acc11[r]; v11 = v11 > 0.0f ? v11 : 0.0f;
    h1bf[(size_t)m0 * HID + col0] = f2bf(v00);
    h1bf[(size_t)m0 * HID + col1] = f2bf(v01);
    h1bf[(size_t)m1 * HID + col0] = f2bf(v10);
    h1bf[(size_t)m1 * HID + col1] = f2bf(v11);
  }
}

// ---------------- Layer 2: gather/mean over h1 + WMMA GEMM + ReLU -> f32 out ----------------
__global__ __launch_bounds__(128) void sage_layer2(
    const u16* __restrict__ h1bf, const int* __restrict__ nb2,
    const int* __restrict__ neigh2, const u16* __restrict__ w2t,
    float* __restrict__ out) {
  __shared__ u16 lds_a[2][32 * 32];
  __shared__ int lds_nb[32];
  __shared__ int lds_ne[32 * SAMP];

  const int tid   = threadIdx.x;
  const int wave  = tid >> 5;
  const int lane  = tid & 31;
  const int mbase = blockIdx.x * 32;

  if (tid < 32) lds_nb[tid] = nb2[mbase + tid];
  for (int i = tid; i < 32 * SAMP; i += 128)
    lds_ne[i] = neigh2[(size_t)(mbase + i / SAMP) * SAMP + (i % SAMP)];

  const int hi   = (lane >= 16) ? 1 : 0;
  const int mrow = lane & 15;
  const int col0 = wave * 32 + (lane & 15);
  const int col1 = col0 + 16;

  auto stage = [&](int buf, int kb) {
    u16* dst = lds_a[buf];
    for (int e = tid; e < 1024; e += 128) {
      int row = e >> 5;
      int kk  = e & 31;
      int k   = kb + kk;
      u16 v;
      if (k < HID) {
        v = h1bf[(size_t)lds_nb[row] * HID + k];   // self part already bf16
      } else {
        int ka = k - HID;
        const int* ne = &lds_ne[row * SAMP];
        float s = 0.0f;
#pragma unroll
        for (int j = 0; j < SAMP; ++j)
          s += bf2f(h1bf[(size_t)ne[j] * HID + ka]);
        v = f2bf(s * (1.0f / SAMP));
      }
      dst[row * 32 + kk] = v;
    }
  };

  v8f acc00 = {}, acc01 = {}, acc10 = {}, acc11 = {};

  __syncthreads();
  stage(0, 0);
  __syncthreads();

  constexpr int NC = K2 / 32;  // 8
  for (int c = 0; c < NC; ++c) {
    const int kb  = c * 32;
    const int cur = c & 1;
    const u16* abuf = lds_a[cur];
    const u16* wchunk = w2t + (size_t)kb * 128;

    FragBF a0, a1, b0, b1;
#pragma unroll
    for (int j = 0; j < 8; ++j) {
      int kj = frag_k(j, hi);
      a0.u[j] = *(const u32*)&abuf[mrow * 32 + kj];
      a1.u[j] = *(const u32*)&abuf[(16 + mrow) * 32 + kj];
      b0.u[j] = *(const u32*)&wchunk[col0 * 32 + kj];
      b1.u[j] = *(const u32*)&wchunk[col1 * 32 + kj];
    }
    acc00 = wmma_bf16(a0, b0, acc00);
    acc01 = wmma_bf16(a0, b1, acc01);
    acc10 = wmma_bf16(a1, b0, acc10);
    acc11 = wmma_bf16(a1, b1, acc11);

    if (c + 1 < NC) stage(cur ^ 1, kb + 32);
    __syncthreads();
  }

#pragma unroll
  for (int r = 0; r < 8; ++r) {
    int m0 = mbase + r + hi * 8;
    int m1 = m0 + 16;
    float v00 = acc00[r]; v00 = v00 > 0.0f ? v00 : 0.0f;
    float v01 = acc01[r]; v01 = v01 > 0.0f ? v01 : 0.0f;
    float v10 = acc10[r]; v10 = v10 > 0.0f ? v10 : 0.0f;
    float v11 = acc11[r]; v11 = v11 > 0.0f ? v11 : 0.0f;
    out[(size_t)m0 * HID + col0] = v00;
    out[(size_t)m0 * HID + col1] = v01;
    out[(size_t)m1 * HID + col0] = v10;
    out[(size_t)m1 * HID + col1] = v11;
  }
}

extern "C" void kernel_launch(void* const* d_in, const int* in_sizes, int n_in,
                              void* d_out, int out_size, void* d_ws, size_t ws_size,
                              hipStream_t stream) {
  const float* raw    = (const float*)d_in[0];  // [100000, 1433]
  const float* W1     = (const float*)d_in[1];  // [128, 2866]
  const float* W2     = (const float*)d_in[2];  // [128, 256]
  const int*   nb1    = (const int*)d_in[3];    // [22528]
  const int*   neigh1 = (const int*)d_in[4];    // [22528, 11]
  const int*   nb2    = (const int*)d_in[5];    // [2048]
  const int*   neigh2 = (const int*)d_in[6];    // [2048, 11]
  float* out = (float*)d_out;                   // [2048, 128]

  char* ws = (char*)d_ws;
  // ws layout (256B aligned):
  //   h1bf : NL1*HID bf16             = 5,767,168 B @ 0
  //   w1t  : 128*K1PAD bf16 (chunked) =   737,280 B @ 5,767,168
  //   w2t  : 128*K2 bf16 (chunked)    =    65,536 B @ 6,504,448
  u16* h1bf = (u16*)(ws);
  u16* w1t  = (u16*)(ws + 5767168);
  u16* w2t  = (u16*)(ws + 6504448);

  prep_w1<<<(128 * K1PAD + 255) / 256, 256, 0, stream>>>(W1, w1t);
  prep_w2<<<(128 * K2 + 255) / 256, 256, 0, stream>>>(W2, w2t);
  sage_layer1<<<NL1 / 32, 128, 0, stream>>>(raw, nb1, neigh1, w1t, h1bf);
  sage_layer2<<<BATCH / 32, 128, 0, stream>>>(h1bf, nb2, neigh2, w2t, out);
}
